// TransformerBlock_20349555048777
// MI455X (gfx1250) — matrix operand order repaired and checked by reading, not compiled, not run on hardware
//
#include <hip/hip_runtime.h>
#include <cstdint>
#include <cstddef>

typedef __attribute__((ext_vector_type(16))) __bf16 v16bf;
typedef __attribute__((ext_vector_type(8)))  float  v8f;
typedef __attribute__((ext_vector_type(4)))  int    v4i;

union FragAB { v16bf v; unsigned int u[8]; };
union FragC  { v8f  v; float f[8]; };

constexpr int S_  = 4096;
constexpr int D_  = 1024;
constexpr int NH_ = 16;
constexpr int HD_ = 64;
constexpr int F_  = 4096;
constexpr float SCALE_ = 0.125f;   // HD^-0.5

#define GAS __attribute__((address_space(1)))
#define LAS __attribute__((address_space(3)))

#if defined(__has_builtin)
#if __has_builtin(__builtin_amdgcn_global_load_async_to_lds_b128)
#define HAVE_ASYNC 1
#endif
#endif
#ifndef HAVE_ASYNC
#define HAVE_ASYNC 0
#endif

// 16B global -> LDS move; async (ASYNCcnt) when available, else sync copy.
__device__ __forceinline__ void ld_b128(const void* g, void* l) {
#if HAVE_ASYNC
  __builtin_amdgcn_global_load_async_to_lds_b128((GAS v4i*)g, (LAS v4i*)l, 0, 0);
#else
  *(uint4*)l = *(const uint4*)g;
#endif
}
__device__ __forceinline__ void async_wait_all() {
#if HAVE_ASYNC
  asm volatile("s_wait_asynccnt 0x0" ::: "memory");
#endif
}
__device__ __forceinline__ void async_wait_4() {
#if HAVE_ASYNC
  asm volatile("s_wait_asynccnt 0x4" ::: "memory");
#endif
}
__device__ __forceinline__ void async_wait_8() {
#if HAVE_ASYNC
  asm volatile("s_wait_asynccnt 0x8" ::: "memory");
#endif
}

__device__ __forceinline__ float gelu_f(float x) {
  float x3 = x * x * x;
  return 0.5f * x * (1.0f + tanhf(0.7978845608028654f * (x + 0.044715f * x3)));
}

// ------------------------------------------------- weight cvt + transpose
// in: K x N (f32)  ->  out: N x K (bf16), tiled via LDS.
__global__ __launch_bounds__(256) void cvt_transpose_kernel(
    const float* __restrict__ in, __bf16* __restrict__ out, int K, int N) {
  __shared__ __bf16 t[32][33];
  int bk = blockIdx.x * 32, bn = blockIdx.y * 32;
  int tx = threadIdx.x & 31, ty = threadIdx.x >> 5;   // 32 x 8
  for (int j = 0; j < 32; j += 8)
    t[ty + j][tx] = (__bf16)in[(size_t)(bk + ty + j) * N + bn + tx];
  __syncthreads();
  for (int j = 0; j < 32; j += 8)
    out[(size_t)(bn + ty + j) * K + bk + tx] = t[tx][ty + j];
}

// ------------------------------------------------- V transpose (per layer)
// qkv (S x 3D, bf16) -> vT (D x S, bf16): vT[c][s] = qkv[s][2D + c]
__global__ __launch_bounds__(256) void transpose_v_kernel(
    const __bf16* __restrict__ qkv, __bf16* __restrict__ vT) {
  __shared__ __bf16 t[32][33];
  int bs = blockIdx.x * 32, bc = blockIdx.y * 32;
  int tx = threadIdx.x & 31, ty = threadIdx.x >> 5;
  for (int j = 0; j < 32; j += 8)
    t[ty + j][tx] = qkv[(size_t)(bs + ty + j) * (3 * D_) + 2 * D_ + bc + tx];
  __syncthreads();
  for (int j = 0; j < 32; j += 8)
    vT[(size_t)(bc + ty + j) * S_ + bs + tx] = t[tx][ty + j];
}

// ---------------------------------------------------------------- layernorm
__global__ __launch_bounds__(256) void ln_kernel(const float* __restrict__ x,
                                                 const float* __restrict__ sc,
                                                 const float* __restrict__ bi,
                                                 __bf16* __restrict__ out) {
  __shared__ float red[256];
  int row = blockIdx.x, tid = threadIdx.x;
  const float* xr = x + (size_t)row * D_;
  float v[4]; float s = 0.f;
  for (int j = 0; j < 4; ++j) { v[j] = xr[tid + j * 256]; s += v[j]; }
  red[tid] = s; __syncthreads();
  for (int st = 128; st > 0; st >>= 1) { if (tid < st) red[tid] += red[tid + st]; __syncthreads(); }
  float mean = red[0] * (1.0f / D_);
  __syncthreads();
  float s2 = 0.f;
  for (int j = 0; j < 4; ++j) { float d = v[j] - mean; s2 += d * d; }
  red[tid] = s2; __syncthreads();
  for (int st = 128; st > 0; st >>= 1) { if (tid < st) red[tid] += red[tid + st]; __syncthreads(); }
  float rstd = rsqrtf(red[0] * (1.0f / D_) + 1e-6f);
  for (int j = 0; j < 4; ++j) {
    int c = tid + j * 256;
    out[(size_t)row * D_ + c] = (__bf16)((v[j] - mean) * rstd * sc[c] + bi[c]);
  }
}

// ---------------------------------------------------------------- WMMA GEMM
// C[M,N] = A[M,K](bf16) @ B (given pre-transposed: Bt[N,K], bf16) + bias.
//   EPI=0: store bf16   EPI=1: +residual, store f32   EPI=2: gelu, store bf16
// Double-buffered LDS, async global->LDS fills overlapped with WMMA.
template <int EPI>
__global__ __launch_bounds__(256) void gemm_bf16_kernel(
    const __bf16* __restrict__ A, const __bf16* __restrict__ Bt,
    const float* __restrict__ bias, const float* __restrict__ resid,
    float* __restrict__ outF, __bf16* __restrict__ outB,
    int M, int N, int K) {
  __shared__ __bf16 As[2][128][40];   // pitch 40 -> 80B rows, 16B aligned
  __shared__ __bf16 Bs[2][128][40];   // Bt tile, row = output col n

  int tid = threadIdx.x;
  int lane = tid & 31, wid = tid >> 5;
  int half = lane >> 4, l15 = lane & 15;
  int wr = wid >> 2, wc = wid & 3;                 // 2x4 wave grid
  int m0 = blockIdx.y * 128, n0 = blockIdx.x * 128;

  int ar = tid >> 1, ac = (tid & 1) * 16;          // coop-load coords (both tiles)

  auto issue = [&](int buf, int k0) {
    const __bf16* gA = A  + (size_t)(m0 + ar) * K + k0 + ac;
    const __bf16* gB = Bt + (size_t)(n0 + ar) * K + k0 + ac;
    ld_b128(gA,     &As[buf][ar][ac]);
    ld_b128(gA + 8, &As[buf][ar][ac + 8]);
    ld_b128(gB,     &Bs[buf][ar][ac]);
    ld_b128(gB + 8, &Bs[buf][ar][ac + 8]);
  };

  FragC acc[4][2];
  for (int mt = 0; mt < 4; ++mt)
    for (int nt = 0; nt < 2; ++nt)
      for (int i = 0; i < 8; ++i) acc[mt][nt].f[i] = 0.0f;

  int ntiles = K / 32;
  issue(0, 0);
  for (int t = 0; t < ntiles; ++t) {
    int cur = t & 1;
    if (t + 1 < ntiles) {
      issue(1 - cur, (t + 1) * 32);
      if (t + 2 < ntiles)   // gfx1250 global_prefetch of tile after next
        __builtin_prefetch(A + (size_t)(m0 + ar) * K + (t + 2) * 32 + ac, 0, 1);
      async_wait_4();       // current tile's 4 async ops retired
    } else {
      async_wait_all();
    }
    __syncthreads();

    FragAB a[4], b[2];
    for (int mt = 0; mt < 4; ++mt) {
      int r = wr * 64 + mt * 16 + l15;
      for (int v = 0; v < 8; ++v) {
        int k = ((v & 3) << 1) + ((v >> 2) << 4) + (half << 3);
        a[mt].u[v] = *(const unsigned int*)&As[cur][r][k];
      }
    }
    for (int nt = 0; nt < 2; ++nt) {
      int c = wc * 32 + nt * 16 + l15;
      for (int v = 0; v < 8; ++v) {
        int k = ((v & 3) << 1) + ((v >> 2) << 4) + (half << 3);
        b[nt].u[v] = *(const unsigned int*)&Bs[cur][c][k];
      }
    }
    for (int mt = 0; mt < 4; ++mt)
      for (int nt = 0; nt < 2; ++nt)
        acc[mt][nt].v = __builtin_amdgcn_wmma_f32_16x16x32_bf16(
            false, a[mt].v, false, b[nt].v, (short)0, acc[mt][nt].v, false, false);
    __syncthreads();   // compute(cur) done before cur is refilled next iter
  }

  for (int mt = 0; mt < 4; ++mt)
    for (int nt = 0; nt < 2; ++nt) {
      int gcol = n0 + wc * 32 + nt * 16 + l15;
      for (int i = 0; i < 8; ++i) {
        int grow = m0 + wr * 64 + mt * 16 + i + 8 * half;
        float v = acc[mt][nt].f[i] + bias[gcol];
        if (EPI == 1) v += resid[(size_t)grow * N + gcol];
        if (EPI == 2) v = gelu_f(v);
        if (EPI == 1) outF[(size_t)grow * N + gcol] = v;
        else          outB[(size_t)grow * N + gcol] = (__bf16)v;
      }
    }
}

// ---------------------------------------------------------------- attention
// Flash-style causal attention, double-buffered K/V tiles.
// qkv rows: [q(1024) k(1024) v(1024)]; vT is pre-transposed V (D x S).
__global__ __launch_bounds__(128) void attn_kernel(const __bf16* __restrict__ qkv,
                                                   const __bf16* __restrict__ vT,
                                                   __bf16* __restrict__ out) {
  __shared__ __bf16 Qs[64][72];
  __shared__ __bf16 Ks[2][64][72];
  __shared__ __bf16 Vts[2][64][72];  // Vts[buf][d][k]
  __shared__ __bf16 Ps[4][16][72];   // per-wave P staging (C-layout -> A-layout)

  int tid = threadIdx.x;
  int w = tid >> 5, lane = tid & 31;
  int half = lane >> 4, l15 = lane & 15;
  int h = blockIdx.y, qi = blockIdx.x;
  int q0 = qi * 64;
  int r = tid >> 1, c0 = (tid & 1) * 32;   // coop-load coords (64 x 64 tiles)

  auto issueKV = [&](int buf, int kt) {
    const __bf16* gk = qkv + (size_t)(kt * 64 + r) * (3 * D_) + D_ + h * HD_ + c0;
    ld_b128(gk,      &Ks[buf][r][c0]);      ld_b128(gk + 8,  &Ks[buf][r][c0 + 8]);
    ld_b128(gk + 16, &Ks[buf][r][c0 + 16]); ld_b128(gk + 24, &Ks[buf][r][c0 + 24]);
    const __bf16* gv = vT + (size_t)(h * HD_ + r) * S_ + kt * 64 + c0;
    ld_b128(gv,      &Vts[buf][r][c0]);      ld_b128(gv + 8,  &Vts[buf][r][c0 + 8]);
    ld_b128(gv + 16, &Vts[buf][r][c0 + 16]); ld_b128(gv + 24, &Vts[buf][r][c0 + 24]);
  };

  { // Q tile 64x64 (completes before first wait<=8: async ops retire in order)
    const __bf16* g = qkv + (size_t)(q0 + r) * (3 * D_) + h * HD_ + c0;
    ld_b128(g, &Qs[r][c0]);      ld_b128(g + 8,  &Qs[r][c0 + 8]);
    ld_b128(g + 16, &Qs[r][c0 + 16]); ld_b128(g + 24, &Qs[r][c0 + 24]);
  }

  float mrow[8], lrow[8];
  FragC O[4];
  for (int i = 0; i < 8; ++i) { mrow[i] = -1e30f; lrow[i] = 0.f; }
  for (int dt = 0; dt < 4; ++dt)
    for (int i = 0; i < 8; ++i) O[dt].f[i] = 0.f;

  issueKV(0, 0);
  for (int kt = 0; kt <= qi; ++kt) {
    int cur = kt & 1;
    if (kt < qi) {
      issueKV(1 - cur, kt + 1);   // overlap next tile's fill with this compute
      async_wait_8();             // current tile's 8 ops (and Q) retired
    } else {
      async_wait_all();
    }
    __syncthreads();

    // S = Q @ K^T  (wave's 16 rows x 64 keys)
    FragC sacc[4];
    for (int nt = 0; nt < 4; ++nt)
      for (int i = 0; i < 8; ++i) sacc[nt].f[i] = 0.f;
    for (int ks = 0; ks < 2; ++ks) {
      FragAB a;
      int arow = w * 16 + l15;
      for (int v = 0; v < 8; ++v) {
        int k = ks * 32 + ((v & 3) << 1) + ((v >> 2) << 4) + (half << 3);
        a.u[v] = *(const unsigned int*)&Qs[arow][k];
      }
      for (int nt = 0; nt < 4; ++nt) {
        FragAB b;
        int bc = nt * 16 + l15;
        for (int v = 0; v < 8; ++v) {
          int k = ks * 32 + ((v & 3) << 1) + ((v >> 2) << 4) + (half << 3);
          b.u[v] = *(const unsigned int*)&Ks[cur][bc][k];
        }
        sacc[nt].v = __builtin_amdgcn_wmma_f32_16x16x32_bf16(
            false, a.v, false, b.v, (short)0, sacc[nt].v, false, false);
      }
    }

    for (int nt = 0; nt < 4; ++nt)
      for (int i = 0; i < 8; ++i) sacc[nt].f[i] *= SCALE_;

    if (kt == qi) {  // wave-uniform branch: mask only the diagonal tile
      for (int nt = 0; nt < 4; ++nt)
        for (int i = 0; i < 8; ++i) {
          int gc = kt * 64 + nt * 16 + l15;
          int gr = q0 + w * 16 + i + 8 * half;
          if (gc > gr) sacc[nt].f[i] = -1e30f;
        }
    }

    // online softmax
    float mx[8];
    for (int i = 0; i < 8; ++i) mx[i] = -1e30f;
    for (int nt = 0; nt < 4; ++nt)
      for (int i = 0; i < 8; ++i) mx[i] = fmaxf(mx[i], sacc[nt].f[i]);
    for (int off = 1; off <= 8; off <<= 1)
      for (int i = 0; i < 8; ++i) mx[i] = fmaxf(mx[i], __shfl_xor(mx[i], off));

    float af[8], rs[8];
    for (int i = 0; i < 8; ++i) {
      float mn = fmaxf(mrow[i], mx[i]);
      af[i] = __expf(mrow[i] - mn);
      mrow[i] = mn; rs[i] = 0.f;
    }
    for (int nt = 0; nt < 4; ++nt)
      for (int i = 0; i < 8; ++i) {
        float p = __expf(sacc[nt].f[i] - mrow[i]);
        rs[i] += p;
        Ps[w][i + 8 * half][nt * 16 + l15] = (__bf16)p;
      }
    for (int off = 1; off <= 8; off <<= 1)
      for (int i = 0; i < 8; ++i) rs[i] += __shfl_xor(rs[i], off);
    for (int i = 0; i < 8; ++i) lrow[i] = lrow[i] * af[i] + rs[i];
    for (int dt = 0; dt < 4; ++dt)
      for (int i = 0; i < 8; ++i) O[dt].f[i] *= af[i];

    // O += P @ V  (same-wave LDS roundtrip re-lays P into A-fragment layout)
    for (int ks = 0; ks < 2; ++ks) {
      FragAB a;
      for (int v = 0; v < 8; ++v) {
        int k = ks * 32 + ((v & 3) << 1) + ((v >> 2) << 4) + (half << 3);
        a.u[v] = *(const unsigned int*)&Ps[w][l15][k];
      }
      for (int dt = 0; dt < 4; ++dt) {
        FragAB b;
        int bc = dt * 16 + l15;
        for (int v = 0; v < 8; ++v) {
          int k = ks * 32 + ((v & 3) << 1) + ((v >> 2) << 4) + (half << 3);
          b.u[v] = *(const unsigned int*)&Vts[cur][bc][k];
        }
        O[dt].v = __builtin_amdgcn_wmma_f32_16x16x32_bf16(
            false, a.v, false, b.v, (short)0, O[dt].v, false, false);
      }
    }
    __syncthreads();   // compute(cur) done before cur is refilled next iter
  }

  for (int dt = 0; dt < 4; ++dt)
    for (int i = 0; i < 8; ++i) {
      int gr = q0 + w * 16 + i + 8 * half;
      int gc = h * HD_ + dt * 16 + l15;
      out[(size_t)gr * D_ + gc] = (__bf16)(O[dt].f[i] * (1.0f / lrow[i]));
    }
}

// ---------------------------------------------------------------- launcher
extern "C" void kernel_launch(void* const* d_in, const int* in_sizes, int n_in,
                              void* d_out, int out_size, void* d_ws, size_t ws_size,
                              hipStream_t stream) {
  const float* x     = (const float*)d_in[0];
  // d_in[1] = mask (causal, handled analytically)
  const float* w_qkv = (const float*)d_in[2];
  const float* b_qkv = (const float*)d_in[3];
  const float* w_out = (const float*)d_in[4];
  const float* b_out = (const float*)d_in[5];
  const float* w_fc1 = (const float*)d_in[6];
  const float* b_fc1 = (const float*)d_in[7];
  const float* w_fc2 = (const float*)d_in[8];
  const float* b_fc2 = (const float*)d_in[9];
  const float* ln1s  = (const float*)d_in[10];
  const float* ln1b  = (const float*)d_in[11];
  const float* ln2s  = (const float*)d_in[12];
  const float* ln2b  = (const float*)d_in[13];
  float* out = (float*)d_out;

  char* p = (char*)d_ws;
  auto carve = [&](size_t bytes) {
    char* r = p; p += (bytes + 255) & ~(size_t)255; return r;
  };
  // weights stored TRANSPOSED (N x K) in bf16
  __bf16* wqkvT = (__bf16*)carve((size_t)3 * D_ * D_ * 2);
  __bf16* woutT = (__bf16*)carve((size_t)D_ * D_ * 2);
  __bf16* wfc1T = (__bf16*)carve((size_t)F_ * D_ * 2);
  __bf16* wfc2T = (__bf16*)carve((size_t)D_ * F_ * 2);
  __bf16* h1_bf = (__bf16*)carve((size_t)S_ * D_ * 2);
  __bf16* qkv_bf= (__bf16*)carve((size_t)S_ * 3 * D_ * 2);
  __bf16* vT    = (__bf16*)carve((size_t)D_ * S_ * 2);
  __bf16* ao_bf = (__bf16*)carve((size_t)S_ * D_ * 2);
  float*  x1    = (float*) carve((size_t)S_ * D_ * 4);
  __bf16* h2_bf = (__bf16*)carve((size_t)S_ * D_ * 2);
  __bf16* fc1_bf= (__bf16*)carve((size_t)S_ * F_ * 2);

  // weight transposes (K x N f32 -> N x K bf16)
  cvt_transpose_kernel<<<dim3(D_ / 32, 3 * D_ / 32), 256, 0, stream>>>(w_qkv, wqkvT, D_, 3 * D_);
  cvt_transpose_kernel<<<dim3(D_ / 32, D_ / 32),     256, 0, stream>>>(w_out, woutT, D_, D_);
  cvt_transpose_kernel<<<dim3(D_ / 32, F_ / 32),     256, 0, stream>>>(w_fc1, wfc1T, D_, F_);
  cvt_transpose_kernel<<<dim3(F_ / 32, D_ / 32),     256, 0, stream>>>(w_fc2, wfc2T, F_, D_);

  // LN1 -> qkv GEMM -> V transpose -> attention -> out-proj (+residual x)
  ln_kernel<<<S_, 256, 0, stream>>>(x, ln1s, ln1b, h1_bf);
  gemm_bf16_kernel<0><<<dim3(3 * D_ / 128, S_ / 128), 256, 0, stream>>>(
      h1_bf, wqkvT, b_qkv, nullptr, nullptr, qkv_bf, S_, 3 * D_, D_);
  transpose_v_kernel<<<dim3(S_ / 32, D_ / 32), 256, 0, stream>>>(qkv_bf, vT);
  attn_kernel<<<dim3(S_ / 64, NH_), 128, 0, stream>>>(qkv_bf, vT, ao_bf);
  gemm_bf16_kernel<1><<<dim3(D_ / 128, S_ / 128), 256, 0, stream>>>(
      ao_bf, woutT, b_out, x, x1, nullptr, S_, D_, D_);

  // LN2 -> fc1(+gelu) -> fc2 (+residual x1) -> d_out
  ln_kernel<<<S_, 256, 0, stream>>>(x1, ln2s, ln2b, h2_bf);
  gemm_bf16_kernel<2><<<dim3(F_ / 128, S_ / 128), 256, 0, stream>>>(
      h2_bf, wfc1T, b_fc1, nullptr, nullptr, fc1_bf, S_, F_, D_);
  gemm_bf16_kernel<1><<<dim3(D_ / 128, S_ / 128), 256, 0, stream>>>(
      fc1_bf, wfc2T, b_fc2, x1, out, nullptr, S_, D_, F_);
}
